// RNNLayer_76235669504204
// MI455X (gfx1250) — matrix operand order repaired and checked by reading, not compiled, not run on hardware
//
#include <hip/hip_runtime.h>
#include <hip/hip_bf16.h>
#include <stdint.h>

// Problem constants (reference: T=2048, B=32, E=512, H=512, V=32000)
#define TT 2048
#define BB 32
#define EE 512
#define HH 512
#define VV 32000

typedef __attribute__((ext_vector_type(16))) _Float16 v16h;
typedef __attribute__((ext_vector_type(8)))  _Float16 v8h;
typedef __attribute__((ext_vector_type(8)))  float    v8f;
typedef __attribute__((ext_vector_type(4)))  uint32_t v4u;
typedef __attribute__((ext_vector_type(8)))  uint32_t v8u;

// LDS (address_space(3)) pointer for async/TDM LDS-address operands.
typedef __attribute__((address_space(3))) void lds_void_t;

__device__ __forceinline__ uint32_t lds_addr_of(const void* p) {
    return (uint32_t)(uintptr_t)(lds_void_t*)p;
}

// CDNA5 async DMA: global -> LDS, 16 bytes per active lane (ASYNCcnt-tracked).
__device__ __forceinline__ void async_b128_to_lds(const void* gaddr, void* lds_ptr) {
    uint32_t l = lds_addr_of(lds_ptr);
    uint64_t g = (uint64_t)(uintptr_t)gaddr;
    asm volatile("global_load_async_to_lds_b128 %0, %1, off"
                 :: "v"(l), "v"(g) : "memory");
}
__device__ __forceinline__ void wait_asynccnt0() {
    asm volatile("s_wait_asynccnt 0x0" ::: "memory");
}

// CDNA5 Tensor Data Mover: 2D tile (f16 elements) global -> LDS.
// D# per CDNA5 ISA: group0 (4 SGPRs) + group1 (8 SGPRs),
// groups 2/3 NULL (2D tensor). TENSORcnt-tracked; EXEC-independent, so the
// caller must issue from a single wave.
__device__ __forceinline__ void tdm_load_2d_f16(uint32_t lds_addr, const void* gaddr,
                                                uint32_t tensor_d0, uint32_t tensor_d1,
                                                uint32_t tile_d0, uint32_t tile_d1,
                                                uint32_t stride0_elems) {
    uint64_t g = (uint64_t)(uintptr_t)gaddr;
    v4u g0;
    g0[0] = 1u;                                               // count=1, user D#
    g0[1] = lds_addr;                                         // lds_addr [63:32]
    g0[2] = (uint32_t)g;                                      // global_addr lo
    g0[3] = ((uint32_t)(g >> 32) & 0x01FFFFFFu) | 0x80000000u; // addr[56:32] | type=2
    v8u g1;
    g1[0] = 1u << 16;                                         // data_size=1 (2 bytes)
    g1[1] = (tensor_d0 & 0xFFFFu) << 16;                      // tensor_dim0[15:0]
    g1[2] = (tensor_d0 >> 16) | ((tensor_d1 & 0xFFFFu) << 16);// dim0 hi | dim1 lo
    g1[3] = (tensor_d1 >> 16) | (tile_d0 << 16);              // dim1 hi | tile_dim0
    g1[4] = tile_d1;                                          // tile_dim1, tile_dim2=0
    g1[5] = stride0_elems;                                    // tensor_dim0_stride lo32
    g1[6] = 0u;                                               // stride0 hi | dim1_stride lo
    g1[7] = 0u;
    asm volatile("tensor_load_to_lds %0, %1" :: "s"(g0), "s"(g1) : "memory");
}

// ---------------------------------------------------------------------------
// Kernel A: f32 -> f16 conversion (emb, W_ih, W_hh), row-major preserved.
// ---------------------------------------------------------------------------
__global__ void f32_to_f16_kernel(const float* __restrict__ in,
                                  _Float16* __restrict__ out, int n) {
    int i = blockIdx.x * blockDim.x + threadIdx.x;
    if (i < n) out[i] = (_Float16)in[i];
}

// ---------------------------------------------------------------------------
// Kernel B: xin[tb, h] = sum_e emb_h[src[tb], e] * wih_h[h, e] + b_ih[h]
// M = 65536 (gathered rows), N = 512, K = 512.  WG = 512 thr (16 waves).
// WG tile: 64 (M) x 512 (N).  Double-buffered K-chunks (32 wide):
//   A (64x32, gathered rows)  : per-lane global_load_async_to_lds_b128
//   B (512x32 slab of W_ih)   : one TDM tensor_load_to_lds descriptor (wave 0)
// Pipeline: wait(kc) -> barrier -> issue(kc+1) -> consume(kc).  1 barrier/chunk.
// ---------------------------------------------------------------------------
__global__ __launch_bounds__(512) void xin_gemm_kernel(
    const int* __restrict__ src,         // [T*B]
    const _Float16* __restrict__ embh,   // [V][E] f16
    const _Float16* __restrict__ wih,    // [H][E] f16 row-major
    const float* __restrict__ bih,       // [H]
    float* __restrict__ xin)             // [T*B][H] f32
{
    __shared__ int srow[64];
    __shared__ __align__(16) _Float16 As[2][64 * 32];    // 2 x 4 KB
    __shared__ __align__(16) _Float16 Bs[2][512 * 32];   // 2 x 32 KB

    const int tid    = threadIdx.x;
    const int wave   = tid >> 5;
    const int lane   = tid & 31;
    const int laneHi = lane >> 4;       // 0/1
    const int lane16 = lane & 15;
    const int mblock = blockIdx.x;      // 0..1023
    const int mt     = wave & 3;        // M-tile within WG tile
    const int ngrp   = wave >> 2;       // 0..3 -> N-tiles ngrp*8 .. +7

    if (tid < 64) srow[tid] = src[mblock * 64 + tid];
    __syncthreads();

    const uint32_t bs_lds[2] = { lds_addr_of(&Bs[0][0]), lds_addr_of(&Bs[1][0]) };

    // --- staging issue helpers -------------------------------------------
    auto issue_chunk = [&](int kc, int buf) {
        // A: 64 rows x 32 halves = 256 x 16B; one async op, threads 0..255.
        if (tid < 256) {
            int r = tid >> 2, seg = tid & 3;
            async_b128_to_lds(embh + (size_t)srow[r] * EE + kc * 32 + seg * 8,
                              &As[buf][r * 32 + seg * 8]);
        }
        // B: one TDM 2D tile: 512 rows x 32 f16, row stride EE.
        if (wave == 0) {
            tdm_load_2d_f16(bs_lds[buf], wih + kc * 32,
                            /*tensor_d0=*/EE, /*tensor_d1=*/HH,
                            /*tile_d0=*/32, /*tile_d1=*/HH,
                            /*stride0=*/EE);
        }
    };

    issue_chunk(0, 0);                   // prologue

    v8f acc[8] = {};
    const int m0 = mt * 16;

    for (int kc = 0; kc < EE / 32; ++kc) {
        const int cur = kc & 1;
        wait_asynccnt0();                          // own async A transfers done
        __builtin_amdgcn_s_wait_tensorcnt(0);      // wave 0's TDM done
        __syncthreads();                           // chunk kc visible to all
        if (kc + 1 < EE / 32) issue_chunk(kc + 1, cur ^ 1);  // DMA under compute

        // A fragment (16x32 f16): lane m = lane16; K runs {0..7},{16..23} (+8 hi)
        const _Float16* ap = &As[cur][(m0 + lane16) * 32];
        v8h alo = *(const v8h*)(ap + (laneHi ? 8 : 0));
        v8h ahi = *(const v8h*)(ap + 16 + (laneHi ? 8 : 0));
        v16h a;
#pragma unroll
        for (int i = 0; i < 8; ++i) { a[i] = alo[i]; a[8 + i] = ahi[i]; }

        // Preload all 8 B fragments (distinct regs -> ds loads overlap WMMAs).
        v16h barr[8];
#pragma unroll
        for (int j = 0; j < 8; ++j) {
            const int ntile = ngrp * 8 + j;
            const _Float16* bp = &Bs[cur][(ntile * 16 + lane16) * 32 + laneHi * 8];
            v8h blo = *(const v8h*)bp;
            v8h bhi = *(const v8h*)(bp + 16);
#pragma unroll
            for (int i = 0; i < 8; ++i) { barr[j][i] = blo[i]; barr[j][8 + i] = bhi[i]; }
        }
#pragma unroll
        for (int j = 0; j < 8; ++j)
            acc[j] = __builtin_amdgcn_wmma_f32_16x16x32_f16(
                false, a, false, barr[j], (short)0, acc[j], false, false);
    }

    // Epilogue: D layout lane n = lane16, vgpr r -> M = r + 8*laneHi
    const size_t rowbase = (size_t)mblock * 64 + m0 + laneHi * 8;
#pragma unroll
    for (int j = 0; j < 8; ++j) {
        const int col = (ngrp * 8 + j) * 16 + lane16;
        const float bv = bih[col];
#pragma unroll
        for (int r = 0; r < 8; ++r)
            xin[(rowbase + r) * HH + col] = acc[j][r] + bv;
    }
}

// ---------------------------------------------------------------------------
// Kernel C: sequential scan  h = tanh(xin_t + h @ W_hh^T + b_hh)
// Batch rows are independent chains -> 2 WGs x 16 batch rows, no cross-WG sync.
// WG = 1024 thr (32 waves); wave w owns N-tile w (cols w*16..w*16+15).
// W_hh f16 fragments register-resident (16 x v16h = 128 VGPRs/wave).
// h (f16) double-buffered in LDS -> 1 barrier/step; A-frag loads software-
// pipelined one K-step ahead of the WMMA chain; xin(t+1) prefetched to WGP$.
// ---------------------------------------------------------------------------
__global__ __launch_bounds__(1024) void rnn_scan_kernel(
    const _Float16* __restrict__ whh,    // [H][H] f16 row-major (row = out col n)
    const float* __restrict__ xin,       // [T*B][H] f32
    const float* __restrict__ bhh,       // [H]
    float* __restrict__ out)             // [T*B*H] outputs ++ [B*H] final
{
    __shared__ __align__(16) _Float16 hbuf[2][16 * HH];  // 32 KB

    const int tid    = threadIdx.x;
    const int wave   = tid >> 5;         // N-tile 0..31
    const int lane   = tid & 31;
    const int laneHi = lane >> 4;
    const int lane16 = lane & 15;
    const int bbase  = blockIdx.x * 16;  // batch-row slice

    // Register-resident W_hh B-fragments for this wave's N-tile, all K.
    v16h bf[16];
#pragma unroll
    for (int k = 0; k < 16; ++k) {
        const _Float16* bp =
            whh + (size_t)(wave * 16 + lane16) * HH + k * 32 + laneHi * 8;
        v8h lo = *(const v8h*)bp;
        v8h hi = *(const v8h*)(bp + 16);
#pragma unroll
        for (int i = 0; i < 8; ++i) { bf[k][i] = lo[i]; bf[k][8 + i] = hi[i]; }
    }
    const int col = wave * 16 + lane16;
    const float bv = bhh[col];

    for (int i = tid; i < 16 * HH; i += 1024) hbuf[0][i] = (_Float16)0.f;
    __syncthreads();

    for (int t = 0; t < TT; ++t) {
        const int cur = t & 1;

        // xin fragment for this step (consumed only in the epilogue) +
        // prefetch of next step's rows into WGP$.
        float xv[8];
        const float* xp = xin + (size_t)(t * BB + bbase + laneHi * 8) * HH + col;
#pragma unroll
        for (int r = 0; r < 8; ++r) {
            xv[r] = xp[(size_t)r * HH];
            __builtin_prefetch(xp + (size_t)BB * HH + (size_t)r * HH, 0, 1);
        }

        // Chained WMMA over K = 512; A-frag load pipelined one step ahead.
        const _Float16* hb = &hbuf[cur][lane16 * HH];
        auto lda = [&](int k) -> v16h {
            const _Float16* ap = hb + k * 32;
            v8h alo = *(const v8h*)(ap + (laneHi ? 8 : 0));
            v8h ahi = *(const v8h*)(ap + 16 + (laneHi ? 8 : 0));
            v16h a;
#pragma unroll
            for (int i = 0; i < 8; ++i) { a[i] = alo[i]; a[8 + i] = ahi[i]; }
            return a;
        };

        v8f acc = {};
        v16h a0 = lda(0);
#pragma unroll
        for (int k = 0; k < 16; ++k) {
            v16h a1 = (k < 15) ? lda(k + 1) : a0;
            acc = __builtin_amdgcn_wmma_f32_16x16x32_f16(
                false, a0, false, bf[k], (short)0, acc, false, false);
            a0 = a1;
        }

        // Epilogue in f32: bias + xin + tanh.
        float hv[8];
#pragma unroll
        for (int r = 0; r < 8; ++r) hv[r] = tanhf(acc[r] + xv[r] + bv);

        // outputs[t, b, :]
        float* op = out + (size_t)(t * BB + bbase + laneHi * 8) * HH + col;
#pragma unroll
        for (int r = 0; r < 8; ++r) op[(size_t)r * HH] = hv[r];

        if (t == TT - 1) {
            float* fp = out + (size_t)TT * BB * HH +
                        (size_t)(bbase + laneHi * 8) * HH + col;
#pragma unroll
            for (int r = 0; r < 8; ++r) fp[(size_t)r * HH] = hv[r];
        }

        // h_new (f16) -> next LDS buffer in A-staging row-major layout.
        _Float16* hn = &hbuf[1 - cur][0];
#pragma unroll
        for (int r = 0; r < 8; ++r)
            hn[(laneHi * 8 + r) * HH + col] = (_Float16)hv[r];

        __syncthreads();   // next step may read buf[1-cur]
    }
}

// ---------------------------------------------------------------------------
// Host-side launcher. Inputs: 0=src(int32 T*B), 1=lengths(scalar, unused),
// 2=emb(f32 V*E), 3=W_ih(f32 H*E), 4=W_hh(f32 H*H), 5=b_ih(f32 H), 6=b_hh(f32 H)
// Output: outputs [T*B*H] f32 then layer_final [B*H] f32.
// ---------------------------------------------------------------------------
extern "C" void kernel_launch(void* const* d_in, const int* in_sizes, int n_in,
                              void* d_out, int out_size, void* d_ws, size_t ws_size,
                              hipStream_t stream) {
    const int*   src = (const int*)d_in[0];
    const float* emb = (const float*)d_in[2];
    const float* wih = (const float*)d_in[3];
    const float* whh = (const float*)d_in[4];
    const float* bih = (const float*)d_in[5];
    const float* bhh = (const float*)d_in[6];
    float*       out = (float*)d_out;

    // Workspace layout (bytes): emb_h | wih_h | whh_h | xin
    char* ws = (char*)d_ws;
    const size_t emb_h_bytes = (size_t)VV * EE * 2;        // 32,768,000
    const size_t w_bytes     = (size_t)HH * EE * 2;        // 524,288
    _Float16* emb_h = (_Float16*)ws;
    _Float16* wih_h = (_Float16*)(ws + emb_h_bytes);
    _Float16* whh_h = (_Float16*)(ws + emb_h_bytes + w_bytes);
    float*    xin   = (float*)(ws + emb_h_bytes + 2 * w_bytes);

    // Phase 0: precision conversion (one-time, bandwidth-trivial).
    {
        int n = VV * EE;
        f32_to_f16_kernel<<<(n + 255) / 256, 256, 0, stream>>>(emb, emb_h, n);
        n = HH * EE;
        f32_to_f16_kernel<<<(n + 255) / 256, 256, 0, stream>>>(wih, wih_h, n);
        n = HH * HH;
        f32_to_f16_kernel<<<(n + 255) / 256, 256, 0, stream>>>(whh, whh_h, n);
    }

    // Phase 1: gathered input projection, 65536x512x512 WMMA GEMM.
    xin_gemm_kernel<<<(TT * BB) / 64, 512, 0, stream>>>(src, emb_h, wih_h, bih, xin);

    // Phase 2: latency-bound recurrence; batch split across 2 WGs.
    rnn_scan_kernel<<<BB / 16, 1024, 0, stream>>>(whh_h, xin, bhh, out);
}
